// Router2_35622458753639
// MI455X (gfx1250) — matrix operand order repaired and checked by reading, not compiled, not run on hardware
//
#include <hip/hip_runtime.h>

typedef __attribute__((ext_vector_type(16))) __bf16 v16bf;
typedef __attribute__((ext_vector_type(8)))  float  v8f;

#define NTOK   32768
#define DIM    2048
#define NEXP   64
#define NITERS 10
#define NEGINF (-__builtin_inff())

// ---------------------------------------------------------------------------
// Kernel 0: split W (f32) into bf16 hi/lo pair, and zero the column offsets c.
// ---------------------------------------------------------------------------
__global__ __launch_bounds__(256) void rtr_split_w(const float* __restrict__ W,
                                                   __bf16* __restrict__ Whi,
                                                   __bf16* __restrict__ Wlo,
                                                   float* __restrict__ c) {
    int i = blockIdx.x * 256 + threadIdx.x;
    if (i < NEXP * DIM) {
        float f = W[i];
        __bf16 h = (__bf16)f;
        Whi[i] = h;
        Wlo[i] = (__bf16)(f - (float)h);
    }
    if (i < NEXP) c[i] = 0.0f;
}

// ---------------------------------------------------------------------------
// Kernel 1: scores = x @ W^T via bf16x3 WMMA (near-f32 accuracy, f32 accum).
// Wave = 16 tokens x 64 experts. Block = 8 waves = 128 tokens. Grid = 256.
// ---------------------------------------------------------------------------
__global__ __launch_bounds__(256) void rtr_gemm(const float* __restrict__ x,
                                                const __bf16* __restrict__ Whi,
                                                const __bf16* __restrict__ Wlo,
                                                float* __restrict__ S) {
    const int wave = threadIdx.x >> 5;
    const int lane = threadIdx.x & 31;
    const int half = lane >> 4;        // 0: lanes 0-15, 1: lanes 16-31
    const int l16  = lane & 15;
    const int tokBase = (blockIdx.x * 8 + wave) * 16;

    const float* xrow = x + (size_t)(tokBase + l16) * DIM;

    v8f acc[4] = {v8f{}, v8f{}, v8f{}, v8f{}};

    for (int k0 = 0; k0 < DIM; k0 += 32) {
        // ---- A tile: 16x32 bf16 layout. Lane needs f32 x[m][k0+16h+0..15];
        //      split each into bf16 hi/lo.
        float fa[16];
        const float4* p0 = (const float4*)(xrow + k0 + 16 * half);
        const float4* p1 = (const float4*)(xrow + k0 + 16 * half + 8);
        *(float4*)&fa[0]  = p0[0];
        *(float4*)&fa[4]  = p0[1];
        *(float4*)&fa[8]  = p1[0];
        *(float4*)&fa[12] = p1[1];
        v16bf ahi, alo;
#pragma unroll
        for (int i = 0; i < 16; ++i) {
            __bf16 h = (__bf16)fa[i];
            ahi[i] = h;
            alo[i] = (__bf16)(fa[i] - (float)h);
        }
        // ---- 4 expert tiles of B (32x16), pre-split W in bf16
#pragma unroll
        for (int t = 0; t < 4; ++t) {
            size_t woff = (size_t)(t * 16 + l16) * DIM + k0 + 16 * half;
            v16bf bhi = *(const v16bf*)(Whi + woff);
            v16bf blo = *(const v16bf*)(Wlo + woff);
            acc[t] = __builtin_amdgcn_wmma_f32_16x16x32_bf16(
                false, ahi, false, bhi, (short)0, acc[t], false, false);
            acc[t] = __builtin_amdgcn_wmma_f32_16x16x32_bf16(
                false, alo, false, bhi, (short)0, acc[t], false, false);
            acc[t] = __builtin_amdgcn_wmma_f32_16x16x32_bf16(
                false, ahi, false, blo, (short)0, acc[t], false, false);
        }
    }
    // ---- C layout: VGPR r holds (M = r + 8*half, N = l16)
#pragma unroll
    for (int t = 0; t < 4; ++t)
#pragma unroll
        for (int r = 0; r < 8; ++r)
            S[(size_t)(tokBase + r + 8 * half) * NEXP + t * 16 + l16] = acc[t][r];
}

// ---------------------------------------------------------------------------
// Kernel 2: one Sinkhorn iteration, fused: per row r_i = LSE_j(S - c_j),
// then online per-column (max, sumexp) of (S - r_i). Block partials out.
// Lane owns columns 2l and 2l+1. ROWS_PER_BLOCK = 128, 8 waves/block.
// ---------------------------------------------------------------------------
#define RPB 128
__global__ __launch_bounds__(256) void rtr_sink_pass(const float* __restrict__ S,
                                                     const float* __restrict__ c,
                                                     float2* __restrict__ partials) {
    const int lane = threadIdx.x & 31;
    const int wave = threadIdx.x >> 5;
    const int j0 = 2 * lane, j1 = 2 * lane + 1;
    const float c0 = c[j0], c1 = c[j1];

    float m0 = NEGINF, s0 = 0.0f, m1 = NEGINF, s1 = 0.0f;
    const int rowBase = blockIdx.x * RPB;

    for (int k = 0; k < RPB / 8; ++k) {
        const int row = rowBase + k * 8 + wave;
        float2 sv = *(const float2*)(S + (size_t)row * NEXP + j0);
        float v0 = sv.x - c0, v1 = sv.y - c1;
        // row LSE over 64 values (butterfly across 32 lanes)
        float m = fmaxf(v0, v1);
        float s = __expf(v0 - m) + __expf(v1 - m);
#pragma unroll
        for (int off = 16; off >= 1; off >>= 1) {
            float mo = __shfl_xor(m, off, 32);
            float so = __shfl_xor(s, off, 32);
            float nm = fmaxf(m, mo);
            s = s * __expf(m - nm) + so * __expf(mo - nm);
            m = nm;
        }
        float r = m + __logf(s);
        // column contributions z = S - r
        float z0 = v0 + c0 - r;
        float z1 = v1 + c1 - r;
        float nm0 = fmaxf(m0, z0);
        s0 = s0 * __expf(m0 - nm0) + __expf(z0 - nm0); m0 = nm0;
        float nm1 = fmaxf(m1, z1);
        s1 = s1 * __expf(m1 - nm1) + __expf(z1 - nm1); m1 = nm1;
    }
    __shared__ float lm[8][NEXP], ls[8][NEXP];
    lm[wave][j0] = m0; ls[wave][j0] = s0;
    lm[wave][j1] = m1; ls[wave][j1] = s1;
    __syncthreads();
    if (threadIdx.x < NEXP) {
        int j = threadIdx.x;
        float M = NEGINF, Ss = 0.0f;
#pragma unroll
        for (int w = 0; w < 8; ++w) {
            float mw = lm[w][j], sw = ls[w][j];
            float nm = fmaxf(M, mw);
            Ss = Ss * __expf(M - nm) + sw * __expf(mw - nm);
            M = nm;
        }
        partials[blockIdx.x * NEXP + j] = make_float2(M, Ss);
    }
}

// ---------------------------------------------------------------------------
// Kernel 3: reduce block partials -> new column offsets c_j (64 threads).
// ---------------------------------------------------------------------------
__global__ void rtr_sink_colreduce(const float2* __restrict__ partials,
                                   int nblk, float* __restrict__ c) {
    int j = threadIdx.x;  // 64 threads
    float M = NEGINF, Ss = 0.0f;
    for (int b = 0; b < nblk; ++b) {
        float2 p = partials[b * NEXP + j];
        float nm = fmaxf(M, p.x);
        Ss = Ss * __expf(M - nm) + p.y * __expf(p.x - nm);
        M = nm;
    }
    c[j] = M + __logf(Ss);
}

// ---------------------------------------------------------------------------
// Kernel 4: top-2 of (S_ij - c_j) per row (r_i is row-constant, drops out),
// gather raw scores. One thread per row.
// ---------------------------------------------------------------------------
__global__ __launch_bounds__(256) void rtr_topk(const float* __restrict__ S,
                                                const float* __restrict__ c,
                                                int* __restrict__ oidx,
                                                float* __restrict__ oc) {
    __shared__ float cs[NEXP];
    if (threadIdx.x < NEXP) cs[threadIdx.x] = c[threadIdx.x];
    __syncthreads();
    int row = blockIdx.x * 256 + threadIdx.x;
    const float* sr = S + (size_t)row * NEXP;
    float v1 = NEGINF, v2 = NEGINF, r1 = 0.0f, r2 = 0.0f;
    int i1 = 0, i2 = 0;
#pragma unroll
    for (int j = 0; j < NEXP; ++j) {
        float raw = sr[j];
        float v = raw - cs[j];
        if (v > v1) { v2 = v1; i2 = i1; r2 = r1; v1 = v; i1 = j; r1 = raw; }
        else if (v > v2) { v2 = v; i2 = j; r2 = raw; }
    }
    oidx[row * 2]     = i1;
    oidx[row * 2 + 1] = i2;
    oc[row * 2]       = r1;
    oc[row * 2 + 1]   = r2;
}

// ---------------------------------------------------------------------------
extern "C" void kernel_launch(void* const* d_in, const int* in_sizes, int n_in,
                              void* d_out, int out_size, void* d_ws, size_t ws_size,
                              hipStream_t stream) {
    const float* x = (const float*)d_in[0];  // [8,4096,2048]
    const float* W = (const float*)d_in[1];  // [64,2048]

    char* ws = (char*)d_ws;
    float*  S        = (float*)ws;                                   // 8 MiB
    __bf16* Whi      = (__bf16*)(ws + 8388608);                      // 256 KiB
    __bf16* Wlo      = (__bf16*)(ws + 8388608 + 262144);             // 256 KiB
    float*  c        = (float*)(ws + 8388608 + 524288);              // 256 B
    float2* partials = (float2*)(ws + 8388608 + 524288 + 256);       // 128 KiB

    const int nblk = NTOK / RPB;  // 256

    rtr_split_w<<<(NEXP * DIM + 255) / 256, 256, 0, stream>>>(W, Whi, Wlo, c);
    rtr_gemm<<<NTOK / 128, 256, 0, stream>>>(x, Whi, Wlo, S);
    for (int it = 0; it < NITERS; ++it) {
        rtr_sink_pass<<<nblk, 256, 0, stream>>>(S, c, partials);
        rtr_sink_colreduce<<<1, NEXP, 0, stream>>>(partials, nblk, c);
    }
    int*   oidx = (int*)d_out;                // 65536 int32 indices
    float* oc   = (float*)d_out + NTOK * 2;   // 65536 f32 gathered scores
    rtr_topk<<<NTOK / 256, 256, 0, stream>>>(S, c, oidx, oc);
}
